// ResidualQuantizationLayer_40029095199350
// MI455X (gfx1250) — compile-verified, matrix operand order checked by reading, not run on hardware
//
#include <hip/hip_runtime.h>

// ---------------- problem constants (match reference) ----------------
static constexpr int Bn = 65536;
static constexpr int Dn = 256;
static constexpr int Kn = 4096;
#define DECAYF 0.99f
#define EPSF   1e-5f

// ---------------- vector types ----------------
typedef __attribute__((ext_vector_type(16))) __bf16 v16bf;
typedef __attribute__((ext_vector_type(8)))  __bf16 v8bf;
typedef __attribute__((ext_vector_type(8)))  float  v8f;

// =====================================================================
// Kernel 0: zero the accumulators in workspace (ws is poisoned 0xAA)
// =====================================================================
__global__ __launch_bounds__(256) void vq_zero_k(float* __restrict__ esumT,
                                                 float* __restrict__ counts,
                                                 float* __restrict__ scalars) {
  size_t i = (size_t)blockIdx.x * blockDim.x + threadIdx.x;
  if (i < (size_t)Kn * Dn) esumT[i] = 0.0f;
  if (i < (size_t)Kn)      counts[i] = 0.0f;
  if (i < 4)               scalars[i] = 0.0f;
}

// =====================================================================
// Kernel 1: transpose codebook to [K][D], split fp32 -> bf16 hi+lo,
// and compute 0.5*||e_k||^2 per column. One block per column k.
// =====================================================================
__global__ __launch_bounds__(256) void vq_prep_embed_k(
    const float* __restrict__ embed,   // [D][K] row-major
    __bf16* __restrict__ eT_hi,        // [K][D]
    __bf16* __restrict__ eT_lo,        // [K][D]
    float* __restrict__ eT_f32,        // [K][D]
    float* __restrict__ halfnorm) {    // [K]
  const int k = blockIdx.x;
  const int d = threadIdx.x;
  float v = embed[(size_t)d * Kn + k];
  size_t o = (size_t)k * Dn + d;
  eT_f32[o] = v;
  __bf16 h = (__bf16)v;
  eT_hi[o] = h;
  eT_lo[o] = (__bf16)(v - (float)h);

  __shared__ float red[256];
  red[d] = v * v;
  __syncthreads();
  for (int s = 128; s > 0; s >>= 1) {
    if (d < s) red[d] += red[d + s];
    __syncthreads();
  }
  if (d == 0) halfnorm[k] = 0.5f * red[0];
}

// =====================================================================
// Kernel 2: fused GEMM + argmin. One block = 16 rows of x; 8 waves each
// sweep a stripe of the 256 column tiles. bf16x3 split WMMA.
// C/D layout (ISA 7.12.2): VGPR j, lanes 0-15 -> M=j, lanes 16-31 -> M=j+8,
// N = lane & 15.
// =====================================================================
__global__ __launch_bounds__(256) void vq_argmin_wmma_k(
    const float* __restrict__ x,         // [B][D]
    const __bf16* __restrict__ eT_hi,    // [K][D]
    const __bf16* __restrict__ eT_lo,    // [K][D]
    const float* __restrict__ halfnorm,  // [K]
    int*   __restrict__ idx_out,         // [B] int
    float* __restrict__ out_idx_f) {     // [B] float (output #0)
  const int lane    = threadIdx.x & 31;
  const int wave    = threadIdx.x >> 5;
  const int halfsel = lane >> 4;      // 0: lanes 0-15, 1: lanes 16-31
  const int lrow    = lane & 15;
  const int r0      = blockIdx.x * 16;

  // ---- build A fragments for all 8 k-chunks (x rows r0..r0+15), hi/lo ----
  // A 16x32 bf16 layout: lanes 0-15 hold K=0..7 (v0-3) and 16..23 (v4-7);
  // lanes 16-31 hold K=8..15 and 24..31 for the same row M=lane-16.
  v16bf a_hi[8], a_lo[8];
  const float* xrow = x + (size_t)(r0 + lrow) * Dn;
#pragma unroll
  for (int c = 0; c < 8; ++c) {
    const int kb0 = 32 * c + halfsel * 8;
    const int kb1 = kb0 + 16;
#pragma unroll
    for (int i = 0; i < 8; ++i) {
      float f0 = xrow[kb0 + i];
      __bf16 h0 = (__bf16)f0;
      a_hi[c][i] = h0;
      a_lo[c][i] = (__bf16)(f0 - (float)h0);
      float f1 = xrow[kb1 + i];
      __bf16 h1 = (__bf16)f1;
      a_hi[c][i + 8] = h1;
      a_lo[c][i + 8] = (__bf16)(f1 - (float)h1);
    }
  }

  float bestVal[8];
  int   bestIdx[8];
#pragma unroll
  for (int j = 0; j < 8; ++j) { bestVal[j] = 3.4e38f; bestIdx[j] = 0; }

  // 256 column tiles total; each wave does 32 (uniform trip count per wave,
  // EXEC stays all-ones around the WMMAs).
  for (int kt = wave; kt < Kn / 16; kt += 8) {
    const int col = kt * 16 + lrow;                 // global codebook column
    const __bf16* bh = eT_hi + (size_t)col * Dn;
    const __bf16* bl = eT_lo + (size_t)col * Dn;
    // hint the next tile of this wave's stripe into cache
    __builtin_prefetch((const void*)(bh + (size_t)128 * Dn), 0, 1);

    v8f acc = {};
#pragma unroll
    for (int c = 0; c < 8; ++c) {
      const int db0 = 32 * c + halfsel * 8;
      v8bf t0 = *(const v8bf*)(bh + db0);
      v8bf t1 = *(const v8bf*)(bh + db0 + 16);
      v8bf u0 = *(const v8bf*)(bl + db0);
      v8bf u1 = *(const v8bf*)(bl + db0 + 16);
      v16bf b_hi, b_lo;
#pragma unroll
      for (int i = 0; i < 8; ++i) {
        b_hi[i] = t0[i]; b_hi[i + 8] = t1[i];
        b_lo[i] = u0[i]; b_lo[i + 8] = u1[i];
      }
      // bf16x3: hi*hi + hi*lo + lo*hi  ~ fp32-accurate dot product
      acc = __builtin_amdgcn_wmma_f32_16x16x32_bf16(false, a_hi[c], false, b_hi,
                                                    (short)0, acc, false, false);
      acc = __builtin_amdgcn_wmma_f32_16x16x32_bf16(false, a_hi[c], false, b_lo,
                                                    (short)0, acc, false, false);
      acc = __builtin_amdgcn_wmma_f32_16x16x32_bf16(false, a_lo[c], false, b_hi,
                                                    (short)0, acc, false, false);
    }
    const float hn = halfnorm[col];
#pragma unroll
    for (int j = 0; j < 8; ++j) {
      float v = hn - acc[j];   // = 0.5||e||^2 - x.e  (argmin-equivalent dist)
      if (v < bestVal[j]) { bestVal[j] = v; bestIdx[j] = col; }
    }
  }

  // reduce across the 16 lanes that share a row (xor within 16-lane halves)
#pragma unroll
  for (int j = 0; j < 8; ++j) {
#pragma unroll
    for (int off = 1; off < 16; off <<= 1) {
      float ov = __shfl_xor(bestVal[j], off, 32);
      int   oi = __shfl_xor(bestIdx[j], off, 32);
      if (ov < bestVal[j]) { bestVal[j] = ov; bestIdx[j] = oi; }
    }
  }

  __shared__ float sVal[8][16];
  __shared__ int   sIdx[8][16];
  if (lrow == 0) {
#pragma unroll
    for (int j = 0; j < 8; ++j) {
      int row = j + 8 * halfsel;
      sVal[wave][row] = bestVal[j];
      sIdx[wave][row] = bestIdx[j];
    }
  }
  __syncthreads();
  if (threadIdx.x < 16) {
    int row = threadIdx.x;
    float bv = sVal[0][row];
    int   bi = sIdx[0][row];
#pragma unroll
    for (int w = 1; w < 8; ++w) {
      if (sVal[w][row] < bv) { bv = sVal[w][row]; bi = sIdx[w][row]; }
    }
    idx_out[r0 + row]   = bi;
    out_idx_f[r0 + row] = (float)bi;
  }
}

// =====================================================================
// Kernel 3: gather quantized rows, quant-loss reduction, EMA scatter-adds.
// One block per row; thread d handles element (row, d).
// =====================================================================
__global__ __launch_bounds__(256) void vq_gather_stats_k(
    const float* __restrict__ x,       // [B][D]
    const float* __restrict__ eT_f32,  // [K][D]
    const int* __restrict__ idx,       // [B]
    float* __restrict__ quant_out,     // [B][D] (output #1)
    float* __restrict__ esumT,         // [K][D] accumulator
    float* __restrict__ counts,        // [K]
    float* __restrict__ scalars) {     // [0]=loss_sum
  const int row = blockIdx.x;
  const int d   = threadIdx.x;
  const int k   = idx[row];
  float q  = eT_f32[(size_t)k * Dn + d];
  float xv = x[(size_t)row * Dn + d];
  quant_out[(size_t)row * Dn + d] = q;  // straight-through output == quantize
  float diff = q - xv;

  __shared__ float red[256];
  red[d] = diff * diff;
  __syncthreads();
  for (int s = 128; s > 0; s >>= 1) {
    if (d < s) red[d] += red[d + s];
    __syncthreads();
  }
  if (d == 0) {
    __hip_atomic_fetch_add(&scalars[0], red[0], __ATOMIC_RELAXED,
                           __HIP_MEMORY_SCOPE_AGENT);
    __hip_atomic_fetch_add(&counts[k], 1.0f, __ATOMIC_RELAXED,
                           __HIP_MEMORY_SCOPE_AGENT);
  }
  __hip_atomic_fetch_add(&esumT[(size_t)k * Dn + d], xv, __ATOMIC_RELAXED,
                         __HIP_MEMORY_SCOPE_AGENT);
}

// =====================================================================
// Kernel 4: EMA cluster sizes + reductions for n and n_small.
// =====================================================================
__global__ __launch_bounds__(256) void vq_ema_cluster_k(
    const float* __restrict__ cluster_size, const float* __restrict__ counts,
    float* __restrict__ ncs_out,  // [K] (output #4)
    float* __restrict__ scalars)  // [1]=n_sum, [2]=n_small
{
  const int k = blockIdx.x * blockDim.x + threadIdx.x;
  float ncs = DECAYF * cluster_size[k] + (1.0f - DECAYF) * counts[k];
  ncs_out[k] = ncs;
  __shared__ float rn[256];
  __shared__ float rs[256];
  rn[threadIdx.x] = ncs;
  rs[threadIdx.x] = (ncs < 1.0f) ? 1.0f : 0.0f;
  __syncthreads();
  for (int s = 128; s > 0; s >>= 1) {
    if (threadIdx.x < s) {
      rn[threadIdx.x] += rn[threadIdx.x + s];
      rs[threadIdx.x] += rs[threadIdx.x + s];
    }
    __syncthreads();
  }
  if (threadIdx.x == 0) {
    __hip_atomic_fetch_add(&scalars[1], rn[0], __ATOMIC_RELAXED,
                           __HIP_MEMORY_SCOPE_AGENT);
    __hip_atomic_fetch_add(&scalars[2], rs[0], __ATOMIC_RELAXED,
                           __HIP_MEMORY_SCOPE_AGENT);
  }
}

// =====================================================================
// Kernel 5: normalized codebook + scalar outputs.
// =====================================================================
__global__ __launch_bounds__(256) void vq_finalize_k(
    const float* __restrict__ embed_avg,  // [D][K]
    const float* __restrict__ esumT,      // [K][D]
    const float* __restrict__ ncs,        // [K]
    const float* __restrict__ scalars,
    float* __restrict__ new_embed,        // [D][K] (output #5)
    float* __restrict__ out_scalars) {    // [0]=n_small, [1]=quant_loss
  size_t i = (size_t)blockIdx.x * blockDim.x + threadIdx.x;  // D*K total
  int k = (int)(i & (Kn - 1));
  int d = (int)(i >> 12);  // i / Kn
  float n  = scalars[1];
  float cs = (ncs[k] + EPSF) / (n + (float)Kn * EPSF) * n;
  float navg = DECAYF * embed_avg[(size_t)d * Kn + k] +
               (1.0f - DECAYF) * esumT[(size_t)k * Dn + d];
  new_embed[(size_t)d * Kn + k] = navg / cs;
  if (i == 0) {
    out_scalars[0] = scalars[2];                          // n_small_clusters
    out_scalars[1] = scalars[0] / (float)((size_t)Bn * Dn);  // mean loss
  }
}

// =====================================================================
extern "C" void kernel_launch(void* const* d_in, const int* in_sizes, int n_in,
                              void* d_out, int out_size, void* d_ws,
                              size_t ws_size, hipStream_t stream) {
  (void)in_sizes; (void)n_in; (void)out_size; (void)ws_size;
  const float* x            = (const float*)d_in[0];  // [B][D]
  const float* embed        = (const float*)d_in[1];  // [D][K]
  const float* cluster_size = (const float*)d_in[2];  // [K]
  const float* embed_avg    = (const float*)d_in[3];  // [D][K]

  // ---- workspace layout ----
  char* ws = (char*)d_ws;
  __bf16* eT_hi   = (__bf16*)ws;            ws += (size_t)Kn * Dn * sizeof(__bf16);
  __bf16* eT_lo   = (__bf16*)ws;            ws += (size_t)Kn * Dn * sizeof(__bf16);
  float*  eT_f32  = (float*)ws;             ws += (size_t)Kn * Dn * sizeof(float);
  float*  halfnorm= (float*)ws;             ws += (size_t)Kn * sizeof(float);
  float*  counts  = (float*)ws;             ws += (size_t)Kn * sizeof(float);
  float*  esumT   = (float*)ws;             ws += (size_t)Kn * Dn * sizeof(float);
  int*    idx_ws  = (int*)ws;               ws += (size_t)Bn * sizeof(int);
  float*  scalars = (float*)ws;             ws += 4 * sizeof(float);

  // ---- output layout (reference tuple order, flattened) ----
  float* out   = (float*)d_out;
  float* o_idx = out;                        // [B]     output
  float* o_q   = o_idx + Bn;                 // [B][D]  quantized
  float* o_sc  = o_q + (size_t)Bn * Dn;      // [2]     n_small, quant_loss
  float* o_ncs = o_sc + 2;                   // [K]     new_cluster_size
  float* o_ne  = o_ncs + Kn;                 // [D][K]  new_embed

  vq_zero_k<<<((size_t)Kn * Dn + 255) / 256, 256, 0, stream>>>(esumT, counts,
                                                               scalars);
  vq_prep_embed_k<<<Kn, 256, 0, stream>>>(embed, eT_hi, eT_lo, eT_f32,
                                          halfnorm);
  vq_argmin_wmma_k<<<Bn / 16, 256, 0, stream>>>(x, eT_hi, eT_lo, halfnorm,
                                                idx_ws, o_idx);
  vq_gather_stats_k<<<Bn, 256, 0, stream>>>(x, eT_f32, idx_ws, o_q, esumT,
                                            counts, scalars);
  vq_ema_cluster_k<<<Kn / 256, 256, 0, stream>>>(cluster_size, counts, o_ncs,
                                                 scalars);
  vq_finalize_k<<<((size_t)Dn * Kn) / 256, 256, 0, stream>>>(
      embed_avg, esumT, o_ncs, scalars, o_ne, o_sc);
}
